// GCNClassifier_14499809591810
// MI455X (gfx1250) — matrix operand order repaired and checked by reading, not compiled, not run on hardware
//
#include <hip/hip_runtime.h>
#include <hip/hip_bf16.h>
#include <math.h>

// ---------------------------------------------------------------------------
// Types for CDNA5 WMMA (wave32): bf16 A/B fragments, f32 accumulator.
// ---------------------------------------------------------------------------
typedef __attribute__((ext_vector_type(16))) __bf16 v16bf;
typedef __attribute__((ext_vector_type(8)))  __bf16 v8bf;
typedef __attribute__((ext_vector_type(8)))  float  v8f;

#define NNODES 100000
#define NEDGES 1600000
#define BN_EPS 1e-5f

// ---------------------------------------------------------------------------
// Small utility kernels
// ---------------------------------------------------------------------------
__global__ void k_zero_f32(float* __restrict__ p, int n) {
    int i = blockIdx.x * blockDim.x + threadIdx.x;
    if (i < n) p[i] = 0.0f;
}

__global__ void k_fill_f32(float* __restrict__ p, float v, int n) {
    int i = blockIdx.x * blockDim.x + threadIdx.x;
    if (i < n) p[i] = v;
}

// f32 -> bf16 elementwise cast
__global__ void k_cast_bf16(const float* __restrict__ src, __bf16* __restrict__ dst, int n) {
    int i = blockIdx.x * blockDim.x + threadIdx.x;
    if (i < n) dst[i] = (__bf16)src[i];
}

// Transpose-cast weights: W[fi][fo] (f32, row-major) -> Wt[fo][fi] (bf16)
__global__ void k_wt_cast(const float* __restrict__ W, __bf16* __restrict__ Wt,
                          int fi, int fo) {
    int i = blockIdx.x * blockDim.x + threadIdx.x;
    if (i < fi * fo) {
        int r = i / fo;      // input-feature index (K)
        int c = i % fo;      // output-feature index
        Wt[(size_t)c * fi + r] = (__bf16)W[i];
    }
}

// ---------------------------------------------------------------------------
// Degree / normalization
// ---------------------------------------------------------------------------
__global__ void k_deg_count(const int* __restrict__ dst, float* __restrict__ deg, int e) {
    int i = blockIdx.x * blockDim.x + threadIdx.x;
    if (i < e) atomicAdd(&deg[dst[i]], 1.0f);
}

__global__ void k_rsqrt_inplace(float* __restrict__ p, int n) {
    int i = blockIdx.x * blockDim.x + threadIdx.x;
    if (i < n) p[i] = rsqrtf(p[i]);
}

// ---------------------------------------------------------------------------
// bf16 WMMA GEMM:  C[n x NOUT] = A[n x K] (bf16) @ Wt^T  (+ bias)
// Wt is stored transposed: Wt[NOUT][K] bf16 so B-fragments are contiguous.
// Block = 256 threads = 8 waves; each wave computes a 16-row strip x NOUT.
// A-fragment layout (ISA 16-bit A 16x32): lane<16 holds K {kb*32+0..7, +16..23},
// lane>=16 holds K {kb*32+8..15, +24..31}, row = m0 + (lane&15).
// B-fragment: lane<16 -> col lane, K {kb*32+0..7, +16..23}; lane>=16 -> col lane-16, K {+8..15, +24..31}.
// C/D layout: VGPR v -> row m0 + v + (lane>=16 ? 8 : 0), col = n0 + (lane&15).
// ---------------------------------------------------------------------------
template<int K, int NOUT>
__global__ __launch_bounds__(256)
void k_gemm_bf16(const __bf16* __restrict__ A, const __bf16* __restrict__ Wt,
                 const float* __restrict__ bias, float* __restrict__ C, int n) {
    constexpr int KB = K / 32;
    constexpr int NT = NOUT / 16;

    const int lane = threadIdx.x & 31;
    const int wave = threadIdx.x >> 5;
    const int m0   = blockIdx.x * 128 + wave * 16;

    int arow = m0 + (lane & 15);
    if (arow > n - 1) arow = n - 1;          // clamp loads for the grid tail (branch-free)

    const __bf16* abase = A + (size_t)arow * K + ((lane & 16) ? 8 : 0);
    const __bf16* wbase = Wt + (size_t)(lane & 15) * K + ((lane & 16) ? 8 : 0);

    v8f acc[NT];
    #pragma unroll
    for (int nt = 0; nt < NT; ++nt)
        #pragma unroll
        for (int v = 0; v < 8; ++v) acc[nt][v] = 0.0f;

    #pragma unroll
    for (int kb = 0; kb < KB; ++kb) {
        v8bf lo = *(const v8bf*)(abase + kb * 32);
        v8bf hi = *(const v8bf*)(abase + kb * 32 + 16);
        v16bf a;
        #pragma unroll
        for (int i = 0; i < 8; ++i) { a[i] = lo[i]; a[i + 8] = hi[i]; }

        #pragma unroll
        for (int nt = 0; nt < NT; ++nt) {
            v8bf blo = *(const v8bf*)(wbase + (size_t)(nt * 16) * K + kb * 32);
            v8bf bhi = *(const v8bf*)(wbase + (size_t)(nt * 16) * K + kb * 32 + 16);
            v16bf b;
            #pragma unroll
            for (int i = 0; i < 8; ++i) { b[i] = blo[i]; b[i + 8] = bhi[i]; }
            acc[nt] = __builtin_amdgcn_wmma_f32_16x16x32_bf16(
                false, a, false, b, (short)0, acc[nt], false, false);
        }
    }

    const int colbase = lane & 15;
    const int rofs    = (lane >> 4) << 3;
    #pragma unroll
    for (int nt = 0; nt < NT; ++nt) {
        const int col = nt * 16 + colbase;
        const float bv = bias ? bias[col] : 0.0f;
        #pragma unroll
        for (int v = 0; v < 8; ++v) {
            const int row = m0 + rofs + v;
            if (row < n) C[(size_t)row * NOUT + col] = acc[nt][v] + bv;
        }
    }
}

// ---------------------------------------------------------------------------
// Edge aggregation: agg[dst] += h[src] * dinv[src]*dinv[dst]   (F = 128)
// One wave per edge; lane handles 4 features -> 4 f32 atomics per lane.
// agg (51 MB) is L2-resident, so atomics stay on-chip.
// ---------------------------------------------------------------------------
__global__ __launch_bounds__(256)
void k_edge_agg(const float* __restrict__ hw, const int* __restrict__ src,
                const int* __restrict__ dst, const float* __restrict__ dinv,
                float* __restrict__ agg, int e) {
    const int wave = (blockIdx.x * blockDim.x + threadIdx.x) >> 5;
    const int lane = threadIdx.x & 31;
    if (wave >= e) return;
    const int s = src[wave];
    const int d = dst[wave];
    const float coef = dinv[s] * dinv[d];
    const float* __restrict__ hr = hw + (size_t)s * 128;
    float* __restrict__ ar = agg + (size_t)d * 128;
    #pragma unroll
    for (int j = 0; j < 4; ++j) {
        const int f = lane + j * 32;
        atomicAdd(ar + f, hr[f] * coef);
    }
}

// conv output = agg + h * dinv^2 + bias ; written in place into hw (F = 128)
__global__ void k_conv_finalize(float* __restrict__ hw, const float* __restrict__ agg,
                                const float* __restrict__ dinv, const float* __restrict__ b,
                                int total) {
    int i = blockIdx.x * blockDim.x + threadIdx.x;
    if (i < total) {
        const int node = i >> 7;
        const int f    = i & 127;
        const float dv = dinv[node];
        hw[i] = agg[i] + hw[i] * dv * dv + b[f];
    }
}

// ---------------------------------------------------------------------------
// BatchNorm: per-column sum & sumsq (blockDim.x == F), then apply+ReLU+bf16.
// ---------------------------------------------------------------------------
__global__ void k_colstats(const float* __restrict__ X, float* __restrict__ stats,
                           int n, int F, int rowsPerBlock) {
    const int c  = threadIdx.x;               // column
    const int r0 = blockIdx.x * rowsPerBlock;
    int r1 = r0 + rowsPerBlock; if (r1 > n) r1 = n;
    float s = 0.0f, s2 = 0.0f;
    for (int r = r0; r < r1; ++r) {
        const float v = X[(size_t)r * F + c];
        s += v; s2 += v * v;
    }
    atomicAdd(&stats[c], s);
    atomicAdd(&stats[F + c], s2);
}

__global__ void k_bn_relu_bf16(const float* __restrict__ X, const float* __restrict__ stats,
                               const float* __restrict__ gamma, const float* __restrict__ beta,
                               __bf16* __restrict__ out, int n, int F) {
    int i = blockIdx.x * blockDim.x + threadIdx.x;
    if (i < n * F) {
        const int c = i % F;
        const float invn = 1.0f / (float)n;
        const float mean = stats[c] * invn;
        const float var  = stats[F + c] * invn - mean * mean;
        const float sc   = gamma[c] * rsqrtf(var + BN_EPS);
        float v = (X[i] - mean) * sc + beta[c];
        v = fmaxf(v, 0.0f);
        out[i] = (__bf16)v;
    }
}

// ---------------------------------------------------------------------------
// Head: out[i] = sigmoid( dot(X[i, 0:128], w) + bo ).  One wave per row.
// ---------------------------------------------------------------------------
__global__ __launch_bounds__(256)
void k_head(const __bf16* __restrict__ X, const __bf16* __restrict__ w,
            const float* __restrict__ bo, float* __restrict__ out, int n) {
    const int row  = (blockIdx.x * blockDim.x + threadIdx.x) >> 5;
    const int lane = threadIdx.x & 31;
    if (row >= n) return;
    const __bf16* __restrict__ r = X + (size_t)row * 128;
    float s = 0.0f;
    #pragma unroll
    for (int j = 0; j < 4; ++j) {
        const int f = lane * 4 + j;
        s += (float)r[f] * (float)w[f];
    }
    #pragma unroll
    for (int o = 16; o > 0; o >>= 1) s += __shfl_xor(s, o, 32);
    if (lane == 0) out[row] = 1.0f / (1.0f + expf(-(s + bo[0])));
}

// ---------------------------------------------------------------------------
// Host-side orchestration
// ---------------------------------------------------------------------------
static inline int cdiv(int a, int b) { return (a + b - 1) / b; }

extern "C" void kernel_launch(void* const* d_in, const int* in_sizes, int n_in,
                              void* d_out, int out_size, void* d_ws, size_t ws_size,
                              hipStream_t stream) {
    (void)in_sizes; (void)n_in; (void)out_size; (void)ws_size;

    const int N = NNODES, E = NEDGES;

    // Inputs (setup_inputs dict order)
    const float* x    = (const float*)d_in[0];
    const int*   ei   = (const int*)  d_in[1];   // [2, E] row-major: src then dst
    const float* Wg1  = (const float*)d_in[2];
    const float* bg1  = (const float*)d_in[3];
    const float* g1   = (const float*)d_in[4];
    const float* be1  = (const float*)d_in[5];
    const float* Wg2  = (const float*)d_in[6];
    const float* bg2  = (const float*)d_in[7];
    const float* g2   = (const float*)d_in[8];
    const float* be2  = (const float*)d_in[9];
    const float* Wm1  = (const float*)d_in[10];
    const float* bm1  = (const float*)d_in[11];
    const float* g3   = (const float*)d_in[12];
    const float* be3  = (const float*)d_in[13];
    const float* Wm2  = (const float*)d_in[14];
    const float* bm2  = (const float*)d_in[15];
    const float* g4   = (const float*)d_in[16];
    const float* be4  = (const float*)d_in[17];
    const float* Wo   = (const float*)d_in[18];
    const float* bo   = (const float*)d_in[19];
    float* out = (float*)d_out;

    const int* src = ei;
    const int* dst = ei + E;

    // Workspace carve-up (all offsets 256B aligned)
    char* ws = (char*)d_ws;
    size_t off = 0;
    float*  B0    = (float*) (ws + off); off += (size_t)N * 256 * 4;  // f32 GEMM out (max width 256)
    float*  B1    = (float*) (ws + off); off += (size_t)N * 128 * 4;  // f32 agg
    __bf16* B2    = (__bf16*)(ws + off); off += (size_t)N * 256 * 2;  // bf16 activations
    float*  dinv  = (float*) (ws + off); off += 400128;               // N f32 (padded to 256B)
    __bf16* Wt1   = (__bf16*)(ws + off); off += 128 * 128 * 2;
    __bf16* Wt2   = (__bf16*)(ws + off); off += 128 * 128 * 2;
    __bf16* Wt3   = (__bf16*)(ws + off); off += 128 * 256 * 2;        // 128 -> 256
    __bf16* Wt4   = (__bf16*)(ws + off); off += 256 * 128 * 2;        // 256 -> 128
    __bf16* Wob   = (__bf16*)(ws + off); off += 256;
    float*  stats = (float*) (ws + off); off += 2048;                 // 512 f32

    // 1) Weight transposed-bf16 casts + Wo cast
    k_wt_cast<<<cdiv(128 * 128, 256), 256, 0, stream>>>(Wg1, Wt1, 128, 128);
    k_wt_cast<<<cdiv(128 * 128, 256), 256, 0, stream>>>(Wg2, Wt2, 128, 128);
    k_wt_cast<<<cdiv(128 * 256, 256), 256, 0, stream>>>(Wm1, Wt3, 128, 256);
    k_wt_cast<<<cdiv(256 * 128, 256), 256, 0, stream>>>(Wm2, Wt4, 256, 128);
    k_cast_bf16<<<1, 128, 0, stream>>>(Wo, Wob, 128);

    // 2) x -> bf16 staging
    k_cast_bf16<<<cdiv(N * 128, 256), 256, 0, stream>>>(x, B2, N * 128);

    // 3) degrees: deg = 1 (self loop) + indegree ; dinv = rsqrt(deg)
    k_fill_f32<<<cdiv(N, 256), 256, 0, stream>>>(dinv, 1.0f, N);
    k_deg_count<<<cdiv(E, 256), 256, 0, stream>>>(dst, dinv, E);
    k_rsqrt_inplace<<<cdiv(N, 256), 256, 0, stream>>>(dinv, N);

    const int gemmBlocks = cdiv(N, 128);

    // ---- GCN block 1 ----
    k_gemm_bf16<128, 128><<<gemmBlocks, 256, 0, stream>>>(B2, Wt1, nullptr, B0, N);
    k_zero_f32<<<cdiv(N * 128, 256), 256, 0, stream>>>(B1, N * 128);
    k_edge_agg<<<cdiv(E, 8), 256, 0, stream>>>(B0, src, dst, dinv, B1, E);
    k_conv_finalize<<<cdiv(N * 128, 256), 256, 0, stream>>>(B0, B1, dinv, bg1, N * 128);
    k_zero_f32<<<2, 256, 0, stream>>>(stats, 512);
    k_colstats<<<cdiv(N, 256), 128, 0, stream>>>(B0, stats, N, 128, 256);
    k_bn_relu_bf16<<<cdiv(N * 128, 256), 256, 0, stream>>>(B0, stats, g1, be1, B2, N, 128);

    // ---- GCN block 2 ----
    k_gemm_bf16<128, 128><<<gemmBlocks, 256, 0, stream>>>(B2, Wt2, nullptr, B0, N);
    k_zero_f32<<<cdiv(N * 128, 256), 256, 0, stream>>>(B1, N * 128);
    k_edge_agg<<<cdiv(E, 8), 256, 0, stream>>>(B0, src, dst, dinv, B1, E);
    k_conv_finalize<<<cdiv(N * 128, 256), 256, 0, stream>>>(B0, B1, dinv, bg2, N * 128);
    k_zero_f32<<<2, 256, 0, stream>>>(stats, 512);
    k_colstats<<<cdiv(N, 256), 128, 0, stream>>>(B0, stats, N, 128, 256);
    k_bn_relu_bf16<<<cdiv(N * 128, 256), 256, 0, stream>>>(B0, stats, g2, be2, B2, N, 128);

    // ---- MLP block 1 (128 -> 256) ----
    k_gemm_bf16<128, 256><<<gemmBlocks, 256, 0, stream>>>(B2, Wt3, bm1, B0, N);
    k_zero_f32<<<2, 256, 0, stream>>>(stats, 512);
    k_colstats<<<cdiv(N, 256), 256, 0, stream>>>(B0, stats, N, 256, 256);
    k_bn_relu_bf16<<<cdiv(N * 256, 256), 256, 0, stream>>>(B0, stats, g3, be3, B2, N, 256);

    // ---- MLP block 2 (256 -> 128) ----
    k_gemm_bf16<256, 128><<<gemmBlocks, 256, 0, stream>>>(B2, Wt4, bm2, B0, N);
    k_zero_f32<<<2, 256, 0, stream>>>(stats, 512);
    k_colstats<<<cdiv(N, 256), 128, 0, stream>>>(B0, stats, N, 128, 256);
    k_bn_relu_bf16<<<cdiv(N * 128, 256), 256, 0, stream>>>(B0, stats, g4, be4, B2, N, 128);

    // ---- Output head ----
    k_head<<<cdiv(N * 32, 256), 256, 0, stream>>>(B2, Wob, bo, out, N);
}
